// CIFLayer_50611894616802
// MI455X (gfx1250) — matrix operand order repaired and checked by reading, not compiled, not run on hardware
//
#include <hip/hip_runtime.h>
#include <hip/hip_bf16.h>
#include <math.h>

typedef __attribute__((ext_vector_type(16))) _Float16 v16h;
typedef __attribute__((ext_vector_type(8)))  _Float16 v8h;
typedef __attribute__((ext_vector_type(8)))  float    v8f;
typedef __attribute__((ext_vector_type(4)))  unsigned int u32x4;
typedef __attribute__((ext_vector_type(8)))  int      i32x8;
typedef __attribute__((ext_vector_type(4)))  int      i32x4;

#define B_   16
#define S_   2048
#define C_   1024
#define H_   1024
#define T_   512
#define T2_  514
#define M_TOTAL (S_*B_)   // 32768 rows of the fused GEMM
#define M_TILE  128
#define XS_STRIDE 1032    // 1024 + 8 halves pad -> 4-bank row stride (conflict-free A loads)
#define WS_ROW    1040    // 1024 + 2x8 halves TDM padding -> 16B-aligned B fragments

// ---------------- workspace layout (bytes) ----------------
static constexpr size_t OFF_W2    = 0;                                   // H_ * 4
static constexpr size_t OFF_W1T   = OFF_W2    + (size_t)H_*4;            // H_*C_*2 (f16 W1^T, (H,C))
static constexpr size_t OFF_ALPHA = OFF_W1T   + (size_t)C_*H_*2;         // B_*S_*4
static constexpr size_t OFF_SCALE = OFF_ALPHA + (size_t)B_*S_*4;         // B_*4 (padded)
static constexpr size_t OFF_RIDX  = OFF_SCALE + 256;
static constexpr size_t OFF_LIDX  = OFF_RIDX  + (size_t)B_*S_*4;
static constexpr size_t OFF_RW    = OFF_LIDX  + (size_t)B_*S_*4;
static constexpr size_t OFF_LW    = OFF_RW    + (size_t)B_*S_*4;
static constexpr size_t OFF_START = OFF_LW    + (size_t)B_*S_*4;
static constexpr size_t OFF_STOP  = OFF_START + (size_t)B_*S_*4;
static constexpr size_t OFF_OUTB  = ((OFF_STOP + (size_t)B_*S_*4) + 255) & ~(size_t)255;
static constexpr size_t OFF_DIFF  = OFF_OUTB  + (size_t)B_*T2_*C_*4;

static constexpr size_t GEMM_LDS_BYTES =
    (size_t)(M_TILE*XS_STRIDE + 16*WS_ROW) * sizeof(_Float16);   // ~291 KB of the 320 KB WGP LDS

// Fast hardware tanh (v_tanh_f32 on gfx1250), guarded fallback.
__device__ __forceinline__ float fast_tanh(float z) {
#if __has_builtin(__builtin_amdgcn_tanhf)
    return __builtin_amdgcn_tanhf(z);
#elif __has_builtin(__builtin_amdgcn_tanh_f32)
    return __builtin_amdgcn_tanh_f32(z);
#else
    return tanhf(z);
#endif
}

// ---------------- kernels ----------------

__global__ void cif_zero(float* __restrict__ p, size_t n) {
    size_t i = (size_t)blockIdx.x * blockDim.x + threadIdx.x;
    size_t stride = (size_t)gridDim.x * blockDim.x;
    for (; i < n; i += stride) p[i] = 0.f;
}

// w2 = g2 * v2 / ||v2||
__global__ void cif_w2norm(const float* __restrict__ v2, const float* __restrict__ g2,
                           float* __restrict__ w2) {
    __shared__ float red[1024];
    int t = threadIdx.x;
    float v = v2[t];
    red[t] = v * v;
    __syncthreads();
    for (int off = 512; off > 0; off >>= 1) {
        if (t < off) red[t] += red[t + off];
        __syncthreads();
    }
    w2[t] = g2[0] * v / sqrtf(red[0]);
}

// W1 (C,H) f32 -> W1^T (H,C) f16: makes the WMMA B-fragment K-direction contiguous
// in memory so the TDM can DMA it straight into LDS.
__global__ void cif_w1_transpose(const float* __restrict__ w1, _Float16* __restrict__ w1t) {
    size_t i = (size_t)blockIdx.x * blockDim.x + threadIdx.x;   // over H*C, k fastest
    int k = (int)(i & (C_ - 1));
    int n = (int)(i >> 10);
    w1t[(size_t)n * C_ + k] = (_Float16)w1[(size_t)k * H_ + n];
}

// B-chunk LDS offset: TDM pads 4 DWORDs (8 halves) after every 256 DWORDs (512 halves),
// so element k of a row lands at k + 8*(k>=512); row stride = WS_ROW halves.
__device__ __forceinline__ int wpad(int k) { return k + ((k >= 512) ? 8 : 0); }

// Fused GEMM:  alpha[b,s] = sigmoid( sum_n tanh( x_row . W1[:,n] + b1[n] ) * w2[n] + b2 )
// 256 threads = 8 waves; each wave owns a 16-row M subtile; WMMA f32_16x16x32_f16.
// W1 chunks are DMA'd into LDS by the Tensor Data Mover (TENSORcnt) when available.
__global__ __launch_bounds__(256)
void cif_gemm_alpha(const float* __restrict__ x, const unsigned char* __restrict__ mask,
                    const float* __restrict__ b1, const float* __restrict__ b2p,
                    const _Float16* __restrict__ w1t, const float* __restrict__ w2,
                    float* __restrict__ alpha) {
    extern __shared__ _Float16 smem[];
    _Float16* xs  = smem;                         // [M_TILE][XS_STRIDE] x tile, f16
    _Float16* wsh = smem + M_TILE * XS_STRIDE;    // [16 cols][WS_ROW]   W1^T chunk

    const int tid  = threadIdx.x;
    const int lane = tid & 31;
    const int wave = tid >> 5;
    const int m0   = blockIdx.x * M_TILE;

    // Stage X tile: coalesced f32 reads, convert to f16 in-register, store to LDS.
    for (int i = tid; i < M_TILE * C_; i += 256) {
        int row = i >> 10;          // / C_
        int k   = i & (C_ - 1);
        xs[row * XS_STRIDE + k] = (_Float16)x[(size_t)(m0 + row) * C_ + k];
    }

    const int g = lane >> 4;        // half-wave group (0/1)
    const int q = lane & 15;
    const int rowbase = wave * 16;
    const _Float16* arow = xs + (rowbase + q) * XS_STRIDE;   // A: lane holds row = lane%16
    const _Float16* bcol = wsh + q * WS_ROW;                 // B: lane holds col = lane%16
    const int klo = g ? 8 : 0;      // A f16 layout: lanes 0-15 K{0..7,16..23}, lanes 16-31 K{8..15,24..31}

    float racc[8];
#pragma unroll
    for (int i = 0; i < 8; ++i) racc[i] = 0.f;

    __syncthreads();

    for (int nt = 0; nt < H_ / 16; ++nt) {
        const int n0 = nt * 16;

#if __has_builtin(__builtin_amdgcn_tensor_load_to_lds)
        // ---- TDM path: one wave issues the DMA, waits on TENSORcnt, then barrier. ----
        if (tid < 32) {
            // D# group 0: count=1 | lds_addr | 57-bit global_addr | type=2 ("image")
            unsigned long long ga = (unsigned long long)(uintptr_t)(w1t + (size_t)n0 * C_);
            u32x4 g0;
            g0[0] = 1u;                                        // count=1, user descriptor
            g0[1] = (unsigned)(uintptr_t)wsh;                  // LDS destination (byte addr)
            g0[2] = (unsigned)(ga & 0xFFFFFFFFu);              // global_addr[31:0]
            g0[3] = (unsigned)((ga >> 32) & 0x1FFFFFFu) | (2u << 30);   // addr[56:32] | type=2
            // D# group 1: data_size=2B, pad_enable, pad_interval=256dw, pad_amount=4dw,
            // tensor_dim0=C (contig K), tensor_dim1=H, tile=1024x16, dim0_stride=C.
            i32x8 g1;
            g1[0] = (1 << 16) | (1 << 20) | (7 << 22) | (3 << 25);
            g1[1] = (int)((unsigned)C_ << 16);                 // tensor_dim0[15:0] @ bits63:48
            g1[2] = (int)((unsigned)H_ << 16);                 // tensor_dim1[15:0] @ bits95:80
            g1[3] = (int)((unsigned)C_ << 16);                 // tile_dim0=1024    @ bits127:112
            g1[4] = 16;                                        // tile_dim1=16      @ bits143:128
            g1[5] = C_;                                        // tensor_dim0_stride @ bits207:160
            g1[6] = 0;
            g1[7] = 0;
            i32x4 gz4 = {0, 0, 0, 0};                          // 2D tensor: groups 2/3 unused
            i32x8 gz8 = {0, 0, 0, 0, 0, 0, 0, 0};              // trailing group (6-arg form)
            __builtin_amdgcn_tensor_load_to_lds(g0, g1, gz4, gz4, gz8, 0);
            __builtin_amdgcn_s_wait_tensorcnt(0);
        }
        __syncthreads();
#else
        // ---- Fallback: cooperative staging with the same padded layout. ----
        for (int i = tid; i < 16 * C_; i += 256) {
            int col = i >> 10;
            int k   = i & (C_ - 1);
            wsh[col * WS_ROW + wpad(k)] = w1t[(size_t)(n0 + col) * C_ + k];
        }
        __syncthreads();
        if (nt + 1 < H_ / 16) {             // prefetch next chunk (global_prefetch_b8)
#pragma unroll
            for (int j = 0; j < 4; ++j)
                __builtin_prefetch(w1t + (size_t)(n0 + 16 + (tid >> 6)) * C_ + ((tid & 63) * 16) + j * 4, 0, 1);
        }
#endif

        // ---- K loop: software-pipelined fragments so DS loads overlap the WMMA pipe. ----
        v8f acc = {};
        v16h a_cur, b_cur;
        {
            v8h alo = *(const v8h*)(arow + klo);
            v8h ahi = *(const v8h*)(arow + 16 + klo);
            v8h blo = *(const v8h*)(bcol + wpad(klo));
            v8h bhi = *(const v8h*)(bcol + wpad(16 + klo));
#pragma unroll
            for (int j = 0; j < 8; ++j) {
                a_cur[j] = alo[j]; a_cur[8 + j] = ahi[j];
                b_cur[j] = blo[j]; b_cur[8 + j] = bhi[j];
            }
        }
#pragma unroll 4
        for (int kc = 0; kc < C_; kc += 32) {
            const int kn = (kc + 32) & (C_ - 1);    // wraps on last iter (harmless reload)
            v16h a_nxt, b_nxt;
            {
                v8h alo = *(const v8h*)(arow + kn + klo);
                v8h ahi = *(const v8h*)(arow + kn + 16 + klo);
                v8h blo = *(const v8h*)(bcol + wpad(kn + klo));
                v8h bhi = *(const v8h*)(bcol + wpad(kn + 16 + klo));
#pragma unroll
                for (int j = 0; j < 8; ++j) {
                    a_nxt[j] = alo[j]; a_nxt[8 + j] = ahi[j];
                    b_nxt[j] = blo[j]; b_nxt[8 + j] = bhi[j];
                }
            }
            acc = __builtin_amdgcn_wmma_f32_16x16x32_f16(
                false, a_cur, false, b_cur, (short)0, acc, false, false);
            a_cur = a_nxt;
            b_cur = b_nxt;
        }

        // Epilogue for this 16-column tile: hw tanh + weight by w2, accumulate per-row.
        const int n = n0 + q;               // C/D layout: lane col = lane%16
        const float b1n = b1[n];
        const float w2n = w2[n];
#pragma unroll
        for (int i = 0; i < 8; ++i)          // VGPR i -> row i + 8*g of the subtile
            racc[i] += fast_tanh(acc[i] + b1n) * w2n;
        __syncthreads();                     // wsh reused next nt
    }

    // Reduce over the 16 lanes sharing a row (xor masks stay within the half-wave group).
#pragma unroll
    for (int m = 1; m < 16; m <<= 1)
#pragma unroll
        for (int i = 0; i < 8; ++i) racc[i] += __shfl_xor(racc[i], m, 32);

    if (q < 8) {
        float val = 0.f;
#pragma unroll
        for (int i = 0; i < 8; ++i) if (q == i) val = racc[i];   // avoid scratch spill
        int mr = m0 + rowbase + q + 8 * g;   // global GEMM row = s*B + b
        int b  = mr & (B_ - 1);
        int s  = mr >> 4;
        float z = val + b2p[0];
        float av = 1.f / (1.f + __expf(-z));
        if (mask[b * S_ + s]) av = 0.f;
        alpha[b * S_ + s] = av;
    }
}

// Per-batch: alpha_sum, scale, and the two small outputs (feat_lengths, alpha_sum).
__global__ void cif_rownorm(const float* __restrict__ alpha, const int* __restrict__ tlen,
                            float* __restrict__ scale, float* __restrict__ out_tail) {
    __shared__ float red[256];
    int b = blockIdx.x, t = threadIdx.x;
    float s = 0.f;
    for (int i = t; i < S_; i += 256) s += alpha[b * S_ + i];
    red[t] = s;
    __syncthreads();
    for (int off = 128; off > 0; off >>= 1) {
        if (t < off) red[t] += red[t + off];
        __syncthreads();
    }
    if (t == 0) {
        float asum = red[0];
        float desired = (float)tlen[b] + 0.0001f;   // BETA = 1
        scale[b] = desired / asum;
        out_tail[b]      = (float)tlen[b];          // feat_lengths
        out_tail[B_ + b] = asum;                    // alpha_sum
    }
}

// Sequential CIF scan per batch: wave32 inclusive scan with carried cumsum / right_idx.
__global__ void cif_scan(const float* __restrict__ alpha, const float* __restrict__ scale,
                         int* __restrict__ ridx, int* __restrict__ lidx,
                         float* __restrict__ rw, float* __restrict__ lw,
                         int* __restrict__ stA, int* __restrict__ spA) {
    int b = blockIdx.x, lane = threadIdx.x;    // blockDim = 32 (one wave)
    float sc = scale[b];
    float carry_c = 0.f;
    int carry_r = 0;
    for (int s0 = 0; s0 < S_; s0 += 32) {
        int s = s0 + lane;
        float a = alpha[b * S_ + s] * sc;
        float v = a;
#pragma unroll
        for (int off = 1; off < 32; off <<= 1) {
            float u = __shfl_up(v, off, 32);
            if (lane >= off) v += u;
        }
        float csum = carry_c + v;
        int r = (int)floorf(csum);
        r = max(0, min(r, T_));
        int l = __shfl_up(r, 1, 32);
        if (lane == 0) l = carry_r;
        int fire  = r - l;
        int extra = max(fire - 1, 0);
        float rwv = (fire > 0) ? (csum - (float)r) : 0.f;
        float lwv = a - rwv - (float)extra;
        int st = min(l + 1, T_ + 1);
        int sp = min(l + 1 + extra, T_ + 1);
        int idx = b * S_ + s;
        ridx[idx] = r; lidx[idx] = l;
        rw[idx] = rwv; lw[idx] = lwv;
        stA[idx] = st; spA[idx] = sp;
        carry_c = __shfl(csum, 31, 32);
        carry_r = __shfl(r, 31, 32);
    }
}

// Scatter: grid (B, 4) x 256 threads; each block owns an exclusive (b, channel-range)
// slice -> no atomics. Index/weight loads are uniform per block (scalar loads).
// out/diff buffers (33.7 MB each) stay resident in the 192 MB L2.
__global__ __launch_bounds__(256)
void cif_scatter(const float* __restrict__ x,
                 const int* __restrict__ ridx, const int* __restrict__ lidx,
                 const float* __restrict__ rw, const float* __restrict__ lw,
                 const int* __restrict__ stA, const int* __restrict__ spA,
                 float* __restrict__ outb, float* __restrict__ diff) {
    int b = blockIdx.x;
    int c = blockIdx.y * blockDim.x + threadIdx.x;   // 0 .. C_-1
    float* ob = outb + (size_t)b * T2_ * C_;
    float* df = diff + (size_t)b * T2_ * C_;
    for (int s = 0; s < S_; ++s) {
        float xv = x[((size_t)s * B_ + b) * C_ + c];
        int idx = b * S_ + s;
        int r = ridx[idx], l = lidx[idx], st = stA[idx], sp = spA[idx];
        float rwv = rw[idx], lwv = lw[idx];
        ob[(size_t)r  * C_ + c] += rwv * xv;
        ob[(size_t)l  * C_ + c] += lwv * xv;
        df[(size_t)st * C_ + c] += xv;        // BETA = 1
        df[(size_t)sp * C_ + c] -= xv;
    }
}

// Fused diff-cumsum + transpose to (T, B, C).
__global__ void cif_final(const float* __restrict__ outb, const float* __restrict__ diff,
                          float* __restrict__ out) {
    int tid = blockIdx.x * blockDim.x + threadIdx.x;   // 0 .. B*C-1
    int b = tid >> 10;
    int c = tid & (C_ - 1);
    const float* ob = outb + (size_t)b * T2_ * C_ + c;
    const float* df = diff + (size_t)b * T2_ * C_ + c;
    float run = 0.f;
    for (int t = 0; t < T_; ++t) {
        run += df[(size_t)t * C_];
        out[((size_t)t * B_ + b) * C_ + c] = ob[(size_t)t * C_] + run;
    }
}

// ---------------- launch ----------------
extern "C" void kernel_launch(void* const* d_in, const int* in_sizes, int n_in,
                              void* d_out, int out_size, void* d_ws, size_t ws_size,
                              hipStream_t stream) {
    const float*         x    = (const float*)d_in[0];
    const unsigned char* mask = (const unsigned char*)d_in[1];
    const int*           tlen = (const int*)d_in[2];
    const float*         w1   = (const float*)d_in[3];
    const float*         b1   = (const float*)d_in[4];
    const float*         v2   = (const float*)d_in[5];
    const float*         g2   = (const float*)d_in[6];
    const float*         b2   = (const float*)d_in[7];

    char* wsp = (char*)d_ws;
    float*    w2v   = (float*)(wsp + OFF_W2);
    _Float16* w1t   = (_Float16*)(wsp + OFF_W1T);
    float*    alpha = (float*)(wsp + OFF_ALPHA);
    float*    scale = (float*)(wsp + OFF_SCALE);
    int*      ridx  = (int*)(wsp + OFF_RIDX);
    int*      lidx  = (int*)(wsp + OFF_LIDX);
    float*    rw    = (float*)(wsp + OFF_RW);
    float*    lw    = (float*)(wsp + OFF_LW);
    int*      stA   = (int*)(wsp + OFF_START);
    int*      spA   = (int*)(wsp + OFF_STOP);
    float*    outb  = (float*)(wsp + OFF_OUTB);
    float*    diff  = (float*)(wsp + OFF_DIFF);
    float*    out   = (float*)d_out;

    // Zero scatter buffers every call (harness replays without re-poisoning).
    cif_zero<<<2048, 256, 0, stream>>>(outb, (size_t)2 * B_ * T2_ * C_);

    cif_w2norm<<<1, 1024, 0, stream>>>(v2, g2, w2v);
    cif_w1_transpose<<<(C_ * H_) / 256, 256, 0, stream>>>(w1, w1t);

    cif_gemm_alpha<<<M_TOTAL / M_TILE, 256, GEMM_LDS_BYTES, stream>>>(
        x, mask, b1, b2, w1t, w2v, alpha);

    cif_rownorm<<<B_, 256, 0, stream>>>(alpha, tlen, scale, out + (size_t)T_ * B_ * C_);
    cif_scan<<<B_, 32, 0, stream>>>(alpha, scale, ridx, lidx, rw, lw, stA, spA);
    cif_scatter<<<dim3(B_, C_ / 256), 256, 0, stream>>>(x, ridx, lidx, rw, lw, stA, spA,
                                                        outb, diff);
    cif_final<<<(B_ * C_) / 256, 256, 0, stream>>>(outb, diff, out);

    (void)in_sizes; (void)n_in; (void)out_size; (void)ws_size;
}
